// DGNN_63290638074458
// MI455X (gfx1250) — matrix operand order repaired and checked by reading, not compiled, not run on hardware
//
#include <hip/hip_runtime.h>
#include <hip/hip_bf16.h>

typedef _Float16 half_t;
typedef __attribute__((ext_vector_type(16))) _Float16 v16h;
typedef __attribute__((ext_vector_type(8)))  _Float16 v8h;
typedef __attribute__((ext_vector_type(4)))  _Float16 v4h;
typedef __attribute__((ext_vector_type(8)))  float    v8f;

#define CH 128

// ---------------- degree / normalization ----------------

__global__ void dgnn_init_one(float* __restrict__ deg, int N) {
    int i = blockIdx.x * blockDim.x + threadIdx.x;
    if (i < N) deg[i] = 1.0f;   // self-loop contributes 1 to every node's degree
}

__global__ void dgnn_deg_count(const int* __restrict__ cols, float* __restrict__ deg, int E) {
    int e = blockIdx.x * blockDim.x + threadIdx.x;
    if (e < E) atomicAdd(&deg[cols[e]], 1.0f);
}

__global__ void dgnn_dinv(float* __restrict__ d, int N) {
    int i = blockIdx.x * blockDim.x + threadIdx.x;
    if (i < N) d[i] = rsqrtf(d[i]);   // deg >= 1 always (self loops)
}

// ---------------- precision prep ----------------

// fp32 -> f16 stream convert (4 elems/thread, b128 in, b64 out)
__global__ void dgnn_cvt_f16(const float* __restrict__ s, half_t* __restrict__ d, int n4) {
    int i = blockIdx.x * blockDim.x + threadIdx.x;
    if (i >= n4) return;
    float4 v = ((const float4*)s)[i];
    v4h h;
    h[0] = (half_t)v.x; h[1] = (half_t)v.y; h[2] = (half_t)v.z; h[3] = (half_t)v.w;
    *(v4h*)(d + (size_t)i * 4) = h;
}

// W[k][n] fp32 row-major -> Wt[n][k] f16 (so B fragments are contiguous in k)
__global__ void dgnn_wt_cvt(const float* __restrict__ W, half_t* __restrict__ Wt) {
    int t = blockIdx.x * blockDim.x + threadIdx.x;   // 0 .. 16383
    int k = t >> 7, n = t & (CH - 1);
    Wt[(size_t)n * CH + k] = (half_t)W[(size_t)k * CH + n];
}

// ---------------- WMMA GEMM: C[N x 128] = A[N x 128] * W[128 x 128] ----------------
// A: f16 row-major. Wt: f16 transposed [n][k]. C: fp32. One wave per 16x16 tile.

__global__ __launch_bounds__(256)
void dgnn_gemm_k128(const half_t* __restrict__ A, const half_t* __restrict__ Wt,
                    float* __restrict__ C, int nrows) {
    int wave = (blockIdx.x * blockDim.x + threadIdx.x) >> 5;
    int lane = threadIdx.x & 31;
    const int tiles_n = CH / 16;            // 8
    int tm = wave / tiles_n;
    int tn = wave % tiles_n;
    if (tm * 16 >= nrows) return;           // wave-uniform: EXEC stays all-ones

    int  l15 = lane & 15;
    bool hi  = lane >= 16;
    int  row  = tm * 16 + l15;              // A-fragment row (M)
    int  colB = tn * 16 + l15;              // B-fragment column (N)

    const half_t* arow = A  + (size_t)row  * CH;
    const half_t* bcol = Wt + (size_t)colB * CH;

    v8f acc = {};
#pragma unroll
    for (int kc = 0; kc < CH; kc += 32) {
        // A 16x32 f16 fragment: halfs[0..7] = A[row][ka..ka+7], halfs[8..15] = A[row][ka+16..ka+23]
        int ka = kc + (hi ? 8 : 0);
        v8h a0 = *(const v8h*)(arow + ka);
        v8h a1 = *(const v8h*)(arow + ka + 16);
        v16h a = __builtin_shufflevector(a0, a1, 0,1,2,3,4,5,6,7,8,9,10,11,12,13,14,15);

        // B 32x16 f16 fragment: halfs[0..7] = Wt[col][kb..kb+7], halfs[8..15] = Wt[col][kb+16..kb+23]
        int kb = kc + (hi ? 8 : 0);
        v8h b0 = *(const v8h*)(bcol + kb);
        v8h b1 = *(const v8h*)(bcol + kb + 16);
        v16h b = __builtin_shufflevector(b0, b1, 0,1,2,3,4,5,6,7,8,9,10,11,12,13,14,15);

        acc = __builtin_amdgcn_wmma_f32_16x16x32_f16(
            /*neg_a=*/false, a, /*neg_b=*/false, b,
            /*c_mod=*/(short)0, acc, /*reuse_a=*/false, /*reuse_b=*/false);
    }

    // D 16x16 f32: VGPR r -> M = r + (hi?8:0), N = lane&15
    int n = tn * 16 + l15;
#pragma unroll
    for (int r = 0; r < 8; ++r) {
        int m = tm * 16 + r + (hi ? 8 : 0);
        C[(size_t)m * CH + n] = acc[r];
    }
}

// ---------------- edge scatter: agg[col] += dinv[row]*dinv[col] * src[row] ----------------
// one wave32 per edge, float4 per lane (128 ch), coalesced gather + f32 atomics

__global__ __launch_bounds__(256)
void dgnn_edge_scatter(const int* __restrict__ rows, const int* __restrict__ cols,
                       const float* __restrict__ dinv, const float* __restrict__ src,
                       float* __restrict__ agg, int E) {
    int tid  = blockIdx.x * blockDim.x + threadIdx.x;
    int e    = tid >> 5;
    int lane = tid & 31;
    if (e >= E) return;
    int r = rows[e], c = cols[e];
    float w = dinv[r] * dinv[c];
    float4 v = ((const float4*)(src + (size_t)r * CH))[lane];
    float* d = agg + (size_t)c * CH + lane * 4;
    atomicAdd(d + 0, w * v.x);
    atomicAdd(d + 1, w * v.y);
    atomicAdd(d + 2, w * v.z);
    atomicAdd(d + 3, w * v.w);
}

// ---------------- fused: self-loop + optional gate + bias + BN statistics ----------------
// block = 128 threads = one channel per thread (coalesced rows); rows strided over grid

__global__ __launch_bounds__(CH)
void dgnn_finish_stats(const float* __restrict__ agg, const float* __restrict__ lin,
                       const float* __restrict__ dinv, const float* __restrict__ bias,
                       const float* __restrict__ gate, float* __restrict__ dst,
                       float* __restrict__ stats, int N) {
    int c = threadIdx.x;
    float bv = bias[c];
    float sum = 0.f, sumsq = 0.f;
    for (int i = blockIdx.x; i < N; i += gridDim.x) {
        float di = dinv[i];
        size_t idx = (size_t)i * CH + c;
        float v = agg[idx] + di * di * lin[idx];   // edge messages + self loop
        if (gate) v *= gate[idx];
        v += bv;
        dst[idx] = v;
        sum += v; sumsq += v * v;
    }
    atomicAdd(&stats[c], sum);
    atomicAdd(&stats[CH + c], sumsq);
}

// batchnorm apply; optionally emits an f16 copy (feeds next WMMA GEMM)
__global__ void dgnn_bn_apply(float* __restrict__ x, half_t* __restrict__ x16,
                              const float* __restrict__ stats,
                              const float* __restrict__ gamma, const float* __restrict__ beta,
                              int N) {
    int idx = blockIdx.x * blockDim.x + threadIdx.x;
    if (idx >= N * CH) return;
    int c = idx & (CH - 1);
    float inv_n = 1.0f / (float)N;
    float mu  = stats[c] * inv_n;
    float var = stats[CH + c] * inv_n - mu * mu;
    float v = (x[idx] - mu) * rsqrtf(var + 1e-5f) * gamma[c] + beta[c];
    x[idx] = v;
    if (x16) x16[idx] = (half_t)v;
}

// ---------------- final projection: res[i] = [h|out][i,:] . out_w + out_b ----------------
// one wave per node, float4 per lane, wave32 shuffle reduction

__global__ __launch_bounds__(256)
void dgnn_final(const float* __restrict__ h, const float* __restrict__ o,
                const float* __restrict__ w, const float* __restrict__ b,
                float* __restrict__ res, int N) {
    int tid  = blockIdx.x * blockDim.x + threadIdx.x;
    int node = tid >> 5;
    int lane = tid & 31;
    if (node >= N) return;
    float4 hv = ((const float4*)(h + (size_t)node * CH))[lane];
    float4 ov = ((const float4*)(o + (size_t)node * CH))[lane];
    float4 w1 = ((const float4*)w)[lane];
    float4 w2 = ((const float4*)(w + CH))[lane];
    float acc = hv.x * w1.x + hv.y * w1.y + hv.z * w1.z + hv.w * w1.w
              + ov.x * w2.x + ov.y * w2.y + ov.z * w2.z + ov.w * w2.w;
#pragma unroll
    for (int off = 16; off > 0; off >>= 1)
        acc += __shfl_down(acc, off, 32);
    if (lane == 0) res[node] = acc + b[0];
}

// ---------------- host-side orchestration ----------------

extern "C" void kernel_launch(void* const* d_in, const int* in_sizes, int n_in,
                              void* d_out, int out_size, void* d_ws, size_t ws_size,
                              hipStream_t stream) {
    const float* x     = (const float*)d_in[0];
    const int*   ei    = (const int*)  d_in[1];
    const float* gcn_w = (const float*)d_in[2];
    const float* gcn_b = (const float*)d_in[3];
    const float* bn1_g = (const float*)d_in[4];
    const float* bn1_b = (const float*)d_in[5];
    const float* dcn_w = (const float*)d_in[6];
    const float* dcn_b = (const float*)d_in[7];
    const float* bn3_g = (const float*)d_in[8];
    const float* bn3_b = (const float*)d_in[9];
    const float* out_w = (const float*)d_in[10];
    const float* out_b = (const float*)d_in[11];

    const int N = in_sizes[0] / CH;      // 50000
    const int E = in_sizes[1] / 2;       // 600000
    const int* rows = ei;                // edge_index[0]
    const int* cols = ei + E;            // edge_index[1]

    // workspace layout
    float*  ws    = (float*)d_ws;
    float*  dinv  = ws;                              // N floats
    float*  stats = dinv + N;                        // 256 floats (sum, sumsq)
    float*  lin   = stats + 256;                     // N*CH f32  (xw, then hw)
    float*  agg   = lin  + (size_t)N * CH;           // N*CH f32
    float*  h     = agg  + (size_t)N * CH;           // N*CH f32
    float*  outb  = h    + (size_t)N * CH;           // N*CH f32
    half_t* ah    = (half_t*)(outb + (size_t)N * CH);// N*CH f16 (x16, then h16)
    half_t* wt1   = ah  + (size_t)N * CH;            // 128*128 f16
    half_t* wt2   = wt1 + CH * CH;                   // 128*128 f16
    float*  res   = (float*)d_out;

    const int T = 256;
    dim3 blk(T);

    // --- degree -> dinv (with self loops) ---
    dgnn_init_one <<<(N + T - 1) / T, blk, 0, stream>>>(dinv, N);
    dgnn_deg_count<<<(E + T - 1) / T, blk, 0, stream>>>(cols, dinv, E);
    dgnn_dinv     <<<(N + T - 1) / T, blk, 0, stream>>>(dinv, N);

    // --- precision prep: f16 weights (transposed) + f16 activations ---
    dgnn_wt_cvt<<<(CH * CH + T - 1) / T, blk, 0, stream>>>(gcn_w, wt1);
    dgnn_wt_cvt<<<(CH * CH + T - 1) / T, blk, 0, stream>>>(dcn_w, wt2);
    const int n4 = N * CH / 4;
    dgnn_cvt_f16<<<(n4 + T - 1) / T, blk, 0, stream>>>(x, ah, n4);

    const int gemm_blocks = (N / 16) * (CH / 16) / 8;   // 8 waves/block
    const size_t nch_bytes = (size_t)N * CH * sizeof(float);

    // --- stage 1: GCNConv + BN ---
    dgnn_gemm_k128<<<gemm_blocks, blk, 0, stream>>>(ah, wt1, lin, N);
    hipMemsetAsync(agg, 0, nch_bytes, stream);
    hipMemsetAsync(stats, 0, 256 * sizeof(float), stream);
    dgnn_edge_scatter<<<((size_t)E * 32 + T - 1) / T, blk, 0, stream>>>(rows, cols, dinv, lin, agg, E);
    dgnn_finish_stats<<<1024, CH, 0, stream>>>(agg, lin, dinv, gcn_b, nullptr, h, stats, N);
    dgnn_bn_apply<<<((size_t)N * CH + T - 1) / T, blk, 0, stream>>>(h, ah, stats, bn1_g, bn1_b, N);

    // --- stage 2: DegGNNConv (gated) + BN ---
    dgnn_gemm_k128<<<gemm_blocks, blk, 0, stream>>>(ah, wt2, lin, N);
    hipMemsetAsync(agg, 0, nch_bytes, stream);
    hipMemsetAsync(stats, 0, 256 * sizeof(float), stream);
    dgnn_edge_scatter<<<((size_t)E * 32 + T - 1) / T, blk, 0, stream>>>(rows, cols, dinv, lin, agg, E);
    dgnn_finish_stats<<<1024, CH, 0, stream>>>(agg, lin, dinv, dcn_b, h, outb, stats, N);
    dgnn_bn_apply<<<((size_t)N * CH + T - 1) / T, blk, 0, stream>>>(outb, nullptr, stats, bn3_g, bn3_b, N);

    // --- final projection ---
    dgnn_final<<<((size_t)N * 32 + T - 1) / T, blk, 0, stream>>>(h, outb, out_w, out_b, res, N);
}
